// CrossAttentionAssociation_52381421142040
// MI455X (gfx1250) — compile-verified, matrix operand order checked by reading, not run on hardware
//
#include <hip/hip_runtime.h>
#include <hip/hip_bf16.h>

#define B_    4
#define ND    512
#define NT    512
#define DM    256
#define DH    128
#define NH    8
#define DK    32
#define RT    8      // row tile for qkv / oln
#define QT    8      // query tile for attention
#define W1S_STRIDE 264   // bf16 elems per row: 256 + 8 pad -> 528 B (16B aligned, banks spread)

typedef __attribute__((ext_vector_type(16))) __bf16 v16bf;
typedef __attribute__((ext_vector_type(8)))  __bf16 v8bf;
typedef __attribute__((ext_vector_type(4)))  __bf16 v4bf;
typedef __attribute__((ext_vector_type(8)))  float  v8f;

// Issue a 32-byte B-tile fetch from LDS: base register + immediate offsets so
// no VALU address math lands after a WMMA (avoids WMMA->VALU WAR hazard NOPs).
#define LDSB_ISSUE(lo, hi, base, imm)                    \
    asm volatile("ds_load_b128 %0, %2 offset:%3\n\t"     \
                 "ds_load_b128 %1, %2 offset:%4"         \
                 : "=v"(lo), "=v"(hi)                    \
                 : "v"(base), "n"(imm), "n"((imm) + 32))

// Wait until at most N DS ops outstanding; ties (lo,hi) through the asm so the
// consuming WMMA can never be scheduled ahead of the wait.
#define LDSB_WAIT2(lo, hi) \
    asm volatile("s_wait_dscnt 0x2" : "+v"(lo), "+v"(hi))
#define LDSB_WAIT0(lo, hi) \
    asm volatile("s_wait_dscnt 0x0" : "+v"(lo), "+v"(hi))

// ---------------------------------------------------------------------------
// 1) QKV projection, 8 rows per block: W streamed once, reused by 8 rows.
//    blockIdx.x = row-tile (0..255), blockIdx.y = which (0=Q,1=K,2=V)
// ---------------------------------------------------------------------------
__global__ __launch_bounds__(256) void qkv_kernel(
    const float* __restrict__ det, const float* __restrict__ trk,
    const float* __restrict__ wq, const float* __restrict__ bq,
    const float* __restrict__ wk, const float* __restrict__ bk,
    const float* __restrict__ wv, const float* __restrict__ bv,
    float* __restrict__ Q, float* __restrict__ K, float* __restrict__ V)
{
    __shared__ float xs[RT][DM];
    const int which = blockIdx.y;
    const int tid   = threadIdx.x;
    const int row0  = blockIdx.x * RT;

    const float* in   = (which == 0) ? det : trk;
    const float* W    = (which == 0) ? wq : (which == 1) ? wk : wv;
    const float* bias = (which == 0) ? bq : (which == 1) ? bk : bv;
    float* outp       = (which == 0) ? Q  : (which == 1) ? K  : V;

    #pragma unroll
    for (int r = 0; r < RT; ++r)
        xs[r][tid] = in[(row0 + r) * DM + tid];
    __syncthreads();

    float acc[RT];
    #pragma unroll
    for (int r = 0; r < RT; ++r) acc[r] = bias[tid];

    const float* wr = W + tid * DM;
    for (int d = 0; d < DM; d += 4) {
        float4 w4 = *(const float4*)(wr + d);
        #pragma unroll
        for (int r = 0; r < RT; ++r)
            acc[r] += w4.x * xs[r][d]     + w4.y * xs[r][d + 1]
                    + w4.z * xs[r][d + 2] + w4.w * xs[r][d + 3];
    }
    const int h = tid >> 5, dk = tid & 31;
    #pragma unroll
    for (int r = 0; r < RT; ++r) {
        const int row = row0 + r;
        const int b = row >> 9, n = row & 511;
        outp[((b * NH + h) * ND + n) * DK + dk] = acc[r];
    }
}

// ---------------------------------------------------------------------------
// 2) Attention, 8 queries per block (K/V streamed once per 8 queries).
//    grid = (ND/QT, NH, B_); 256 threads = 8 waves.
// ---------------------------------------------------------------------------
__global__ __launch_bounds__(256) void attn_kernel(
    const float* __restrict__ Q, const float* __restrict__ K,
    const float* __restrict__ V, float* __restrict__ ctx)
{
    __shared__ float qs[QT][DK];
    __shared__ float ps[QT][NT];       // 16 KB
    __shared__ float wred[QT][8];
    __shared__ float mrow[QT];
    __shared__ float dinv[QT];

    const int qbase = blockIdx.x * QT;
    const int h = blockIdx.y, b = blockIdx.z;
    const int tid = threadIdx.x;
    const int wv = tid >> 5, l = tid & 31;

    const float* Qb = Q + ((b * NH + h) * ND + qbase) * DK;
    const float* Kb = K + (b * NH + h) * NT * DK;
    const float* Vb = V + (b * NH + h) * NT * DK;

    qs[wv][l] = Qb[wv * DK + l];       // 8 waves x 32 lanes = full Q tile
    __syncthreads();

    const float scale = 0.17677669529663687f;  // 1/sqrt(32)
    float sreg[2][QT];
    float sm[QT];
    #pragma unroll
    for (int q = 0; q < QT; ++q) sm[q] = -1e30f;

    #pragma unroll
    for (int i = 0; i < 2; ++i) {
        const int k = tid + i * 256;
        const float* Kr = Kb + k * DK;
        float4 kv[8];
        #pragma unroll
        for (int j = 0; j < 8; ++j) kv[j] = *(const float4*)(Kr + 4 * j);
        #pragma unroll
        for (int q = 0; q < QT; ++q) {
            float s = 0.f;
            #pragma unroll
            for (int j = 0; j < 8; ++j)
                s += kv[j].x * qs[q][4*j]   + kv[j].y * qs[q][4*j+1]
                   + kv[j].z * qs[q][4*j+2] + kv[j].w * qs[q][4*j+3];
            s *= scale;
            sreg[i][q] = s;
            sm[q] = fmaxf(sm[q], s);
        }
    }

    // row max: wave shfl + cross-wave via LDS
    #pragma unroll
    for (int q = 0; q < QT; ++q) {
        float m = sm[q];
        m = fmaxf(m, __shfl_xor(m, 1, 32));
        m = fmaxf(m, __shfl_xor(m, 2, 32));
        m = fmaxf(m, __shfl_xor(m, 4, 32));
        m = fmaxf(m, __shfl_xor(m, 8, 32));
        m = fmaxf(m, __shfl_xor(m, 16, 32));
        if (l == 0) wred[q][wv] = m;
    }
    __syncthreads();
    if (tid < QT) {
        float m = wred[tid][0];
        #pragma unroll
        for (int j = 1; j < 8; ++j) m = fmaxf(m, wred[tid][j]);
        mrow[tid] = m;
    }
    __syncthreads();

    float ls[QT];
    #pragma unroll
    for (int q = 0; q < QT; ++q) {
        const float m = mrow[q];
        const float p0 = __expf(sreg[0][q] - m);
        const float p1 = __expf(sreg[1][q] - m);
        ps[q][tid]       = p0;
        ps[q][tid + 256] = p1;
        ls[q] = p0 + p1;
    }
    #pragma unroll
    for (int q = 0; q < QT; ++q) {
        float s = ls[q];
        s += __shfl_xor(s, 1, 32);
        s += __shfl_xor(s, 2, 32);
        s += __shfl_xor(s, 4, 32);
        s += __shfl_xor(s, 8, 32);
        s += __shfl_xor(s, 16, 32);
        if (l == 0) wred[q][wv] = s;
    }
    __syncthreads();
    if (tid < QT) {
        float s = 0.f;
        #pragma unroll
        for (int j = 0; j < 8; ++j) s += wred[tid][j];
        dinv[tid] = 1.f / s;
    }
    __syncthreads();

    // ctx: wave wv owns query wv, lane l owns dk = l
    float c = 0.f;
    for (int k = 0; k < NT; ++k)
        c += ps[wv][k] * Vb[k * DK + l];
    ctx[(b * ND + qbase + wv) * DM + h * DK + l] = c * dinv[wv];
}

// ---------------------------------------------------------------------------
// 3) Output projection + residual + LayerNorm, 8 rows per block.
// ---------------------------------------------------------------------------
__global__ __launch_bounds__(256) void oln_kernel(
    const float* __restrict__ det, const float* __restrict__ ctx,
    const float* __restrict__ wo, const float* __restrict__ bo,
    const float* __restrict__ g, const float* __restrict__ beta,
    float* __restrict__ xn)
{
    __shared__ float cs[RT][DM];
    __shared__ float xb[RT][DM];
    __shared__ float mus[RT];
    __shared__ float invs[RT];
    const int tid  = threadIdx.x;
    const int row0 = blockIdx.x * RT;

    #pragma unroll
    for (int r = 0; r < RT; ++r)
        cs[r][tid] = ctx[(row0 + r) * DM + tid];
    __syncthreads();

    float acc[RT];
    #pragma unroll
    for (int r = 0; r < RT; ++r) acc[r] = bo[tid];

    const float* wr = wo + tid * DM;
    for (int d = 0; d < DM; d += 4) {
        float4 w4 = *(const float4*)(wr + d);
        #pragma unroll
        for (int r = 0; r < RT; ++r)
            acc[r] += w4.x * cs[r][d]     + w4.y * cs[r][d + 1]
                    + w4.z * cs[r][d + 2] + w4.w * cs[r][d + 3];
    }
    #pragma unroll
    for (int r = 0; r < RT; ++r)
        xb[r][tid] = det[(row0 + r) * DM + tid] + acc[r];
    __syncthreads();

    // wave w reduces row w (sum, sumsq)
    const int w = tid >> 5, l = tid & 31;
    float s = 0.f, s2 = 0.f;
    #pragma unroll
    for (int i = 0; i < DM / 32; ++i) {
        const float v = xb[w][l + 32 * i];
        s += v; s2 += v * v;
    }
    s  += __shfl_xor(s, 1, 32);  s2 += __shfl_xor(s2, 1, 32);
    s  += __shfl_xor(s, 2, 32);  s2 += __shfl_xor(s2, 2, 32);
    s  += __shfl_xor(s, 4, 32);  s2 += __shfl_xor(s2, 4, 32);
    s  += __shfl_xor(s, 8, 32);  s2 += __shfl_xor(s2, 8, 32);
    s  += __shfl_xor(s, 16, 32); s2 += __shfl_xor(s2, 16, 32);
    if (l == 0) {
        const float mu = s * (1.f / DM);
        mus[w]  = mu;
        invs[w] = rsqrtf(s2 * (1.f / DM) - mu * mu + 1e-5f);
    }
    __syncthreads();

    #pragma unroll
    for (int r = 0; r < RT; ++r)
        xn[(row0 + r) * DM + tid] = (xb[r][tid] - mus[r]) * invs[r] * g[tid] + beta[tid];
}

// ---------------------------------------------------------------------------
// 4) tracks fp32 -> bf16 (row-major [B][NT][DM])
// ---------------------------------------------------------------------------
__global__ void cvt_kernel(const float* __restrict__ in, __bf16* __restrict__ outb, int n4)
{
    const int i = blockIdx.x * blockDim.x + threadIdx.x;
    if (i < n4) {
        float4 v = ((const float4*)in)[i];
        v4bf p;
        p.x = (__bf16)v.x; p.y = (__bf16)v.y; p.z = (__bf16)v.z; p.w = (__bf16)v.w;
        ((v4bf*)outb)[i] = p;
    }
}

// ---------------------------------------------------------------------------
// 5) Association head: per (b,n):
//      W1n = w1 * xn_row (bf16, LDS)
//      H   = tracks[b] (512x256) @ W1n^T (256x128)  via v_wmma_f32_16x16x32_bf16
//      out[t] = sigmoid( relu(H + b1) @ w2 + b2 )
// ---------------------------------------------------------------------------
__global__ __launch_bounds__(256) void assoc_kernel(
    const __bf16* __restrict__ trkbf,  // [B][NT][DM] bf16
    const float* __restrict__ xn,      // [B][ND][DM]
    const float* __restrict__ w1,      // [DH][DM]
    const float* __restrict__ b1,      // [DH]
    const float* __restrict__ w2,      // [DH]
    const float* __restrict__ b2,      // [1]
    float* __restrict__ out)           // [B][ND][NT]
{
    __shared__ __bf16 w1s[DH * W1S_STRIDE];   // 67,584 bytes

    const int n   = blockIdx.x;
    const int b   = blockIdx.y;
    const int tid = threadIdx.x;

    const float* xrow = xn + (b * ND + n) * DM;

    // Build W1n = w1 ⊙ xn_row into LDS (bf16, padded rows).
    for (int idx = tid; idx < DH * (DM / 4); idx += 256) {
        const int j = idx >> 6;           // DM/4 == 64
        const int d = (idx & 63) << 2;
        const float4 wv = *(const float4*)(w1 + j * DM + d);
        const float4 xv = *(const float4*)(xrow + d);
        v4bf p;
        p.x = (__bf16)(wv.x * xv.x);
        p.y = (__bf16)(wv.y * xv.y);
        p.z = (__bf16)(wv.z * xv.z);
        p.w = (__bf16)(wv.w * xv.w);
        *(v4bf*)(&w1s[j * W1S_STRIDE + d]) = p;
    }
    __syncthreads();

    const int wave  = tid >> 5;
    const int lane  = tid & 31;
    const int lhalf = lane & 15;
    const int khalf = (lane >> 4) << 3;   // 0 for lanes 0-15, 8 for lanes 16-31

    float b1v[8], w2v[8];
    #pragma unroll
    for (int nt = 0; nt < 8; ++nt) {
        b1v[nt] = b1[nt * 16 + lhalf];
        w2v[nt] = w2[nt * 16 + lhalf];
    }
    const float bias2 = b2[0];

    // Per-lane LDS byte address of each B-tile row base (generic LDS pointer:
    // low 32 bits are the LDS byte offset per the aperture mapping).
    unsigned bbase[8];
    #pragma unroll
    for (int nt = 0; nt < 8; ++nt)
        bbase[nt] = (unsigned)(unsigned long long)(const void*)
                        &w1s[(nt * 16 + lhalf) * W1S_STRIDE + khalf];

    const __bf16* trk = trkbf + b * NT * DM;
    float* orow = out + (b * ND + n) * NT;

    for (int mi = 0; mi < 4; ++mi) {
        const int mbase = (wave * 4 + mi) * 16;
        const __bf16* arowp = trk + (mbase + lhalf) * DM + khalf;

        v8f acc[8];
        #pragma unroll
        for (int nt = 0; nt < 8; ++nt)
            #pragma unroll
            for (int e = 0; e < 8; ++e) acc[nt][e] = 0.f;

        // Double-buffered B pipeline across the 64 (k0,nt) tiles of this mi.
        v8bf blo[2], bhi[2];
        LDSB_ISSUE(blo[0], bhi[0], bbase[0], 0);   // preload tile (k0=0, nt=0)

        #pragma unroll
        for (int k0 = 0; k0 < DM; k0 += 32) {
            // A tile: 16x32 bf16 (lane<16 -> K {k0..k0+7, k0+16..23}; lane>=16 -> +8),
            // two contiguous 16B chunks per lane, reused across all 8 N-tiles.
            union { v16bf v; v8bf h[2]; } A;
            A.h[0] = *(const v8bf*)(arowp + k0);
            A.h[1] = *(const v8bf*)(arowp + k0 + 16);

            #pragma unroll
            for (int nt = 0; nt < 8; ++nt) {
                const int t    = (k0 >> 2) + nt;        // (k0/32)*8 + nt
                const int buf  = t & 1;
                const bool last = (k0 == DM - 32) && (nt == 7);

                if (!last) {
                    const int nnt = (nt + 1) & 7;
                    const int nk0 = (nt == 7) ? (k0 + 32) : k0;
                    LDSB_ISSUE(blo[buf ^ 1], bhi[buf ^ 1], bbase[nnt], nk0 * 2);
                    LDSB_WAIT2(blo[buf], bhi[buf]);     // tile t's data is home
                } else {
                    LDSB_WAIT0(blo[buf], bhi[buf]);
                }

                union { v16bf v; v8bf h[2]; } Bm;
                Bm.h[0] = blo[buf];
                Bm.h[1] = bhi[buf];

                acc[nt] = __builtin_amdgcn_wmma_f32_16x16x32_bf16(
                    false, A.v, false, Bm.v, (short)0, acc[nt], false, false);
            }
        }

        // Epilogue: C/D layout: VGPR r, lane l -> M = r + 8*(l>>4), N = l&15.
        #pragma unroll
        for (int r = 0; r < 8; ++r) {
            float s = 0.f;
            #pragma unroll
            for (int nt = 0; nt < 8; ++nt) {
                const float hval = acc[nt][r] + b1v[nt];
                s += fmaxf(hval, 0.f) * w2v[nt];
            }
            // reduce across the 16 lanes of this half (N dimension)
            s += __shfl_xor(s, 1, 32);
            s += __shfl_xor(s, 2, 32);
            s += __shfl_xor(s, 4, 32);
            s += __shfl_xor(s, 8, 32);
            if (lhalf == 0) {
                const int t = mbase + r + khalf;   // khalf == 8*(lane>>4)
                const float logit = s + bias2;
                orow[t] = 1.f / (1.f + __expf(-logit));
            }
        }
    }
}

// ---------------------------------------------------------------------------
// Launcher
// ---------------------------------------------------------------------------
extern "C" void kernel_launch(void* const* d_in, const int* in_sizes, int n_in,
                              void* d_out, int out_size, void* d_ws, size_t ws_size,
                              hipStream_t stream)
{
    const float* det = (const float*)d_in[0];
    const float* trk = (const float*)d_in[1];
    const float* wq  = (const float*)d_in[2];
    const float* bq  = (const float*)d_in[3];
    const float* wk  = (const float*)d_in[4];
    const float* bk  = (const float*)d_in[5];
    const float* wv  = (const float*)d_in[6];
    const float* bv  = (const float*)d_in[7];
    const float* wo  = (const float*)d_in[8];
    const float* bo  = (const float*)d_in[9];
    const float* lng = (const float*)d_in[10];
    const float* lnb = (const float*)d_in[11];
    const float* w1  = (const float*)d_in[12];
    const float* b1  = (const float*)d_in[13];
    const float* w2  = (const float*)d_in[14];
    const float* b2  = (const float*)d_in[15];
    float* out = (float*)d_out;

    const size_t rowsD = (size_t)B_ * ND * DM;  // 524288 floats
    float* Q   = (float*)d_ws;
    float* K   = Q + rowsD;
    float* V   = K + rowsD;
    float* CTX = V + rowsD;
    float* XN  = CTX + rowsD;
    __bf16* TRKB = (__bf16*)(XN + rowsD);       // 1 MB of bf16

    qkv_kernel<<<dim3(B_ * ND / RT, 3), 256, 0, stream>>>(det, trk, wq, bq, wk, bk, wv, bv, Q, K, V);
    cvt_kernel<<<dim3((B_ * NT * DM / 4 + 255) / 256), 256, 0, stream>>>(trk, TRKB, B_ * NT * DM / 4);
    attn_kernel<<<dim3(ND / QT, NH, B_), 256, 0, stream>>>(Q, K, V, CTX);
    oln_kernel<<<dim3(B_ * ND / RT), 256, 0, stream>>>(det, CTX, wo, bo, lng, lnb, XN);
    assoc_kernel<<<dim3(ND, B_), 256, 0, stream>>>(TRKB, XN, w1, b1, w2, b2, out);
}
